// CrossAttentionGraphBlock_20409684591115
// MI455X (gfx1250) — matrix operand order repaired and checked by reading, not compiled, not run on hardware
//
#include <hip/hip_runtime.h>
#include <hip/hip_bf16.h>
#include <stdint.h>
#include <stddef.h>

// ---------------------------------------------------------------------------
// CDNA5 (gfx1250) wave32 WMMA types & helpers
// ---------------------------------------------------------------------------
typedef __attribute__((ext_vector_type(16))) _Float16 v16h;
typedef __attribute__((ext_vector_type(8)))  _Float16 v8h;
typedef __attribute__((ext_vector_type(8)))  float    v8f;
typedef __attribute__((ext_vector_type(4)))  int      v4i;

union F16x16 { v16h v; v8h h[2]; };

__device__ __forceinline__ v8f wmma_f16f32(v16h a, v16h b, v8f c) {
  // D = A(16x32 f16) * B(32x16 f16) + C(16x16 f32)
  return __builtin_amdgcn_wmma_f32_16x16x32_f16(
      /*neg_a=*/false, a, /*neg_b=*/false, b,
      /*c_mod=*/(short)0, c, /*reuse_a=*/false, /*reuse_b=*/false);
}

// Async global->LDS DMA path (CDNA5 ASYNCcnt). Guarded so the file still
// compiles (with a software-pipelined fallback) if the builtin is absent.
#if defined(__AMDGCN__) && __has_builtin(__builtin_amdgcn_global_load_async_to_lds_b128)
#define USE_ASYNC_LDS 1
#define GEMM_NSTAGE 2
#else
#define USE_ASYNC_LDS 0
#define GEMM_NSTAGE 1
#endif

__device__ __forceinline__ void async_wait0() {
#if USE_ASYNC_LDS
#if __has_builtin(__builtin_amdgcn_s_wait_asynccnt)
  __builtin_amdgcn_s_wait_asynccnt(0);
#else
  asm volatile("s_wait_asynccnt 0x0" ::: "memory");
#endif
#endif
}

#define CB   16
#define CNQ  512
#define CNK  1024
#define CD   768
#define CL   512
#define CH   12
#define CDH  64

// ---------------------------------------------------------------------------
// Elementwise f32 -> f16 convert
// ---------------------------------------------------------------------------
__global__ __launch_bounds__(256) void cvt_f16_kernel(const float* __restrict__ in,
                                                      _Float16* __restrict__ out,
                                                      long long n) {
  long long id = (long long)blockIdx.x * 256 + threadIdx.x;
  if (id < n) out[id] = (_Float16)in[id];
}

// Convert + transpose: W[K][N] f32 -> Wt[N][K] f16
__global__ __launch_bounds__(256) void cvt_tr_kernel(const float* __restrict__ W,
                                                     _Float16* __restrict__ Wt,
                                                     int K, int N) {
  int id = blockIdx.x * 256 + threadIdx.x;
  if (id >= K * N) return;
  int n = id / K, k = id % K;
  Wt[id] = (_Float16)W[(size_t)k * N + n];
}

// Fused bias: beff[n] = sum_d b1[d]*W2[d][n] + b2[n]   (tiny: 768x768)
__global__ __launch_bounds__(256) void fuse_b_kernel(const float* __restrict__ b1,
                                                     const float* __restrict__ W2,
                                                     const float* __restrict__ b2,
                                                     float* __restrict__ beff,
                                                     int Dm, int N) {
  int n = blockIdx.x * 256 + threadIdx.x;
  if (n >= N) return;
  float a = b2[n];
  for (int d = 0; d < Dm; ++d) a += b1[d] * W2[(size_t)d * N + n];
  beff[n] = a;
}

// vh [B*NK, D] f16 -> vhT [B, H, DH, NK] f16 (key-contiguous for B-fragments)
__global__ __launch_bounds__(256) void vt_kernel(const _Float16* __restrict__ vh,
                                                 _Float16* __restrict__ vhT) {
  long long id = (long long)blockIdx.x * 256 + threadIdx.x;
  const long long total = (long long)CB * CH * CDH * CNK;
  if (id >= total) return;
  int key = (int)(id & (CNK - 1));
  long long rest = id >> 10;          // (b*H + h)*DH + dh
  int dh = (int)(rest & (CDH - 1));
  long long bh = rest >> 6;           // b*H + h
  int h = (int)(bh % CH);
  int b = (int)(bh / CH);
  vhT[id] = vh[((size_t)(b * CNK + key)) * CD + h * CDH + dh];
}

// ---------------------------------------------------------------------------
// Tiled WMMA GEMM:  out[M,N] = A[M,K](f16) @ Bt[N,K]^T(f16) + bias
// 256 threads = 8 waves; block tile 128x128, k-step 32; wave tile 64x32.
// Double-buffered LDS fed by async DMA (ASYNCcnt) when available, else
// software-pipelined register staging.
// EP: 0 = f16 store, 1 = f32 store, 2 = f32 leaky_relu(x)+res,
//     3 = f16 store TRANSPOSED (out[col*M+row], no bias) for weight fusion.
// ---------------------------------------------------------------------------
template <int EP>
__global__ __launch_bounds__(256) void gemm_kernel(
    const _Float16* __restrict__ A, const _Float16* __restrict__ Bt,
    const float* __restrict__ bias, const float* __restrict__ res,
    _Float16* __restrict__ outh, float* __restrict__ outf,
    int M, int N, int K) {
  __shared__ _Float16 As[GEMM_NSTAGE][128 * 40];  // stride 40 halves: conflict-free
  __shared__ _Float16 Bs[GEMM_NSTAGE][128 * 40];

  const int t = threadIdx.x;
  const int lane = t & 31, wid = t >> 5;
  const int l16 = lane & 15, half = lane >> 4;
  const int wm = wid >> 2, wn = wid & 3;
  const int mBase = blockIdx.y * 128, nBase = blockIdx.x * 128;

  const int srow = t >> 1, sseg = (t & 1) * 16;
  const _Float16* aG = A + (size_t)(mBase + srow) * K + sseg;
  const _Float16* bG = Bt + (size_t)(nBase + srow) * K + sseg;

  v8f acc[4][2];
  v8f vzero = {};
#pragma unroll
  for (int mi = 0; mi < 4; ++mi)
#pragma unroll
    for (int ni = 0; ni < 2; ++ni) acc[mi][ni] = vzero;

  auto compute_tile = [&](const _Float16* Ab, const _Float16* Bb) {
    F16x16 af[4], bf[2];
#pragma unroll
    for (int mi = 0; mi < 4; ++mi) {
      const _Float16* p = &Ab[(wm * 64 + mi * 16 + l16) * 40 + half * 8];
      af[mi].h[0] = *(const v8h*)p;          // K = base..base+7
      af[mi].h[1] = *(const v8h*)(p + 16);   // K = base+16..base+23
    }
#pragma unroll
    for (int ni = 0; ni < 2; ++ni) {
      const _Float16* p = &Bb[(wn * 32 + ni * 16 + l16) * 40 + half * 8];
      bf[ni].h[0] = *(const v8h*)p;          // K = base..base+7
      bf[ni].h[1] = *(const v8h*)(p + 16);   // K = base+16..base+23
    }
#pragma unroll
    for (int mi = 0; mi < 4; ++mi)
#pragma unroll
      for (int ni = 0; ni < 2; ++ni)
        acc[mi][ni] = wmma_f16f32(af[mi].v, bf[ni].v, acc[mi][ni]);
  };

  const int nk = K >> 5;

#if USE_ASYNC_LDS
  auto stage = [&](int s) {
    // builtin signature (per hipcc diagnostic): (int4* src, int4* dst, Ii, Ii)
    __builtin_amdgcn_global_load_async_to_lds_b128(
        (v4i*)aG,       (v4i*)&As[s][srow * 40 + sseg],     0, 0);
    __builtin_amdgcn_global_load_async_to_lds_b128(
        (v4i*)(aG + 8), (v4i*)&As[s][srow * 40 + sseg + 8], 0, 0);
    __builtin_amdgcn_global_load_async_to_lds_b128(
        (v4i*)bG,       (v4i*)&Bs[s][srow * 40 + sseg],     0, 0);
    __builtin_amdgcn_global_load_async_to_lds_b128(
        (v4i*)(bG + 8), (v4i*)&Bs[s][srow * 40 + sseg + 8], 0, 0);
    aG += 32; bG += 32;
  };
  stage(0);
  async_wait0();
  __syncthreads();
  for (int kt = 0; kt < nk; ++kt) {
    const int cur = kt & 1;
    if (kt + 1 < nk) stage(cur ^ 1);      // DMA next tile under the WMMAs
    compute_tile(&As[cur][0], &Bs[cur][0]);
    async_wait0();
    __syncthreads();
  }
#else
  // software pipeline: prefetch next tile's global loads into registers
  v8h ra0 = *(const v8h*)(aG), ra1 = *(const v8h*)(aG + 8);
  v8h rb0 = *(const v8h*)(bG), rb1 = *(const v8h*)(bG + 8);
  aG += 32; bG += 32;
  for (int kt = 0; kt < nk; ++kt) {
    __syncthreads();
    *(v8h*)&As[0][srow * 40 + sseg]     = ra0;
    *(v8h*)&As[0][srow * 40 + sseg + 8] = ra1;
    *(v8h*)&Bs[0][srow * 40 + sseg]     = rb0;
    *(v8h*)&Bs[0][srow * 40 + sseg + 8] = rb1;
    __syncthreads();
    if (kt + 1 < nk) {
      ra0 = *(const v8h*)(aG); ra1 = *(const v8h*)(aG + 8);
      rb0 = *(const v8h*)(bG); rb1 = *(const v8h*)(bG + 8);
      aG += 32; bG += 32;
    }
    compute_tile(&As[0][0], &Bs[0][0]);
  }
#endif

  // Epilogue. C-layout: lane -> col = base + (lane&15); VGPR r -> row r + 8*half
#pragma unroll
  for (int ni = 0; ni < 2; ++ni) {
    int col = nBase + wn * 32 + ni * 16 + l16;
    float bv = (EP == 3) ? 0.f : bias[col];
#pragma unroll
    for (int mi = 0; mi < 4; ++mi) {
#pragma unroll
      for (int r = 0; r < 8; ++r) {
        int row = mBase + wm * 64 + mi * 16 + r + 8 * half;
        float v = acc[mi][ni][r] + bv;
        if (EP == 0) {
          outh[(size_t)row * N + col] = (_Float16)v;
        } else if (EP == 3) {
          outh[(size_t)col * M + row] = (_Float16)v;   // transposed store
        } else if (EP == 1) {
          outf[(size_t)row * N + col] = v;
        } else {
          v = v > 0.f ? v : 0.01f * v;                 // leaky_relu
          outf[(size_t)row * N + col] = v + res[(size_t)row * N + col];
        }
      }
    }
  }
}

// ---------------------------------------------------------------------------
// Flash-attention: one wave per (b, h, 16-query tile). Online softmax.
// Q fragments register-resident; K/V B-fragments loaded contiguously from
// global (kh is dh-contiguous; vhT is key-contiguous). P re-layout via a
// per-wave LDS bounce guarded by s_wait_dscnt.
// ---------------------------------------------------------------------------
__global__ __launch_bounds__(256) void attn_kernel(
    const _Float16* __restrict__ qh, const _Float16* __restrict__ kh,
    const _Float16* __restrict__ vhT, const unsigned char* __restrict__ mask,
    _Float16* __restrict__ ctx) {
  __shared__ _Float16 plds[8][16 * 40];   // per-wave P bounce, padded stride

  const int lane = threadIdx.x & 31;
  const int wid  = threadIdx.x >> 5;
  const int gw = blockIdx.x * 8 + wid;    // 6144 waves total
  const int qt = gw & 31;                 // NQ/16 = 32 tiles
  const int h  = (gw >> 5) % CH;
  const int b  = gw / (32 * CH);
  const int l16 = lane & 15, half = lane >> 4;
  const int kb8 = half * 8;

  // Q fragments (A-layout), two k-steps covering DH=64
  const _Float16* qrow = qh + ((size_t)(b * CNQ + qt * 16 + l16)) * CD + h * CDH;
  F16x16 aq[2];
  aq[0].h[0] = *(const v8h*)(qrow + kb8);
  aq[0].h[1] = *(const v8h*)(qrow + kb8 + 16);
  aq[1].h[0] = *(const v8h*)(qrow + 32 + kb8);
  aq[1].h[1] = *(const v8h*)(qrow + 32 + kb8 + 16);

  float mrow[8], lrow[8];
  v8f acc[4];
  v8f vzero = {};
#pragma unroll
  for (int r = 0; r < 8; ++r) { mrow[r] = -1e30f; lrow[r] = 0.f; }
#pragma unroll
  for (int n = 0; n < 4; ++n) acc[n] = vzero;

  const unsigned char* mp = mask + (size_t)b * CNK;
  const _Float16* khb = kh + ((size_t)b * CNK) * CD + h * CDH;
  const _Float16* vtb = vhT + (((size_t)(b * CH + h)) * CDH) * CNK;

  for (int kc = 0; kc < CNK; kc += 32) {
    v8f s[2];
#pragma unroll
    for (int g = 0; g < 2; ++g) {
      int key = kc + g * 16 + l16;
      const _Float16* kp = khb + (size_t)key * CD + half * 8;
      F16x16 b0, b1;
      b0.h[0] = *(const v8h*)(kp);        // dh = base..base+7
      b0.h[1] = *(const v8h*)(kp + 16);   // dh = base+16..base+23
      b1.h[0] = *(const v8h*)(kp + 32);
      b1.h[1] = *(const v8h*)(kp + 48);
      v8f z = vzero;
      z = wmma_f16f32(aq[0].v, b0.v, z);
      z = wmma_f16f32(aq[1].v, b1.v, z);
      bool dead = (mp[key] != 0);
#pragma unroll
      for (int r = 0; r < 8; ++r) z[r] = dead ? -1e30f : z[r] * 0.125f;
      s[g] = z;
    }

    // row max over 32 keys (columns striped across a half-wave)
    float rmax[8], rsum[8], alpha[8];
#pragma unroll
    for (int r = 0; r < 8; ++r) rmax[r] = fmaxf(s[0][r], s[1][r]);
#pragma unroll
    for (int o = 8; o >= 1; o >>= 1)
#pragma unroll
      for (int r = 0; r < 8; ++r)
        rmax[r] = fmaxf(rmax[r], __shfl_xor(rmax[r], o, 32));
#pragma unroll
    for (int r = 0; r < 8; ++r) {
      float mn = fmaxf(mrow[r], rmax[r]);
      alpha[r] = __expf(mrow[r] - mn);
      mrow[r] = mn;
      rsum[r] = 0.f;
    }
    // P = exp(S - m); bounce C-layout -> A-layout through LDS
#pragma unroll
    for (int g = 0; g < 2; ++g)
#pragma unroll
      for (int r = 0; r < 8; ++r) {
        float p = __expf(s[g][r] - mrow[r]);
        rsum[r] += p;
        plds[wid][(r + 8 * half) * 40 + g * 16 + l16] = (_Float16)p;
      }
#pragma unroll
    for (int o = 8; o >= 1; o >>= 1)
#pragma unroll
      for (int r = 0; r < 8; ++r) rsum[r] += __shfl_xor(rsum[r], o, 32);
#pragma unroll
    for (int r = 0; r < 8; ++r) lrow[r] = lrow[r] * alpha[r] + rsum[r];
#pragma unroll
    for (int n = 0; n < 4; ++n)
#pragma unroll
      for (int r = 0; r < 8; ++r) acc[n][r] *= alpha[r];

    // cross-lane LDS RAW: wait for this wave's ds stores to land
    asm volatile("s_wait_dscnt 0x0" ::: "memory");

    F16x16 pf;
    const _Float16* pl = &plds[wid][l16 * 40 + kb8];
    pf.h[0] = *(const v8h*)pl;
    pf.h[1] = *(const v8h*)(pl + 16);

#pragma unroll
    for (int n = 0; n < 4; ++n) {
      const _Float16* vp = vtb + (size_t)(n * 16 + l16) * CNK + kc + half * 8;
      F16x16 vf;
      vf.h[0] = *(const v8h*)(vp);
      vf.h[1] = *(const v8h*)(vp + 16);
      acc[n] = wmma_f16f32(pf.v, vf.v, acc[n]);
    }
  }

  // ctx[b, q, h*64 + dh] = acc / l
#pragma unroll
  for (int n = 0; n < 4; ++n)
#pragma unroll
    for (int r = 0; r < 8; ++r) {
      int row = b * CNQ + qt * 16 + r + 8 * half;
      ctx[(size_t)row * CD + h * CDH + n * 16 + l16] =
          (_Float16)(acc[n][r] / lrow[r]);
    }
}

// ---------------------------------------------------------------------------
// LayerNorm over last dim (768). One block per row. Optional residual add
// and optional f16 shadow copy (feeds the next GEMM).
// ---------------------------------------------------------------------------
__global__ __launch_bounds__(256) void ln_kernel(
    const float* __restrict__ in, const float* __restrict__ res,
    const float* __restrict__ g, const float* __restrict__ bb,
    float* __restrict__ outf, _Float16* __restrict__ outh, int N) {
  const int row = blockIdx.x;
  const float* x = in + (size_t)row * N;
  float v[3];
#pragma unroll
  for (int i = 0; i < 3; ++i) {
    int c = threadIdx.x + i * 256;
    v[i] = x[c];
    if (res) v[i] += res[(size_t)row * N + c];
  }
  __shared__ float red[256];
  float s = v[0] + v[1] + v[2];
  red[threadIdx.x] = s;
  __syncthreads();
  for (int o = 128; o > 0; o >>= 1) {
    if (threadIdx.x < o) red[threadIdx.x] += red[threadIdx.x + o];
    __syncthreads();
  }
  float mean = red[0] * (1.0f / 768.0f);
  __syncthreads();
  float sq = 0.f;
#pragma unroll
  for (int i = 0; i < 3; ++i) { float d = v[i] - mean; sq += d * d; }
  red[threadIdx.x] = sq;
  __syncthreads();
  for (int o = 128; o > 0; o >>= 1) {
    if (threadIdx.x < o) red[threadIdx.x] += red[threadIdx.x + o];
    __syncthreads();
  }
  float rstd = rsqrtf(red[0] * (1.0f / 768.0f) + 1e-5f);
#pragma unroll
  for (int i = 0; i < 3; ++i) {
    int c = threadIdx.x + i * 256;
    float o = (v[i] - mean) * rstd * g[c] + bb[c];
    outf[(size_t)row * N + c] = o;
    if (outh) outh[(size_t)row * N + c] = (_Float16)o;
  }
}

// ---------------------------------------------------------------------------
// Orchestration
// ---------------------------------------------------------------------------
extern "C" void kernel_launch(void* const* d_in, const int* in_sizes, int n_in,
                              void* d_out, int out_size, void* d_ws, size_t ws_size,
                              hipStream_t stream) {
  (void)in_sizes; (void)n_in; (void)out_size; (void)ws_size;

  const float* gn   = (const float*)d_in[0];   // [B, NQ, D]
  const float* cond = (const float*)d_in[1];   // [B, NK, L]
  const float* qW   = (const float*)d_in[2];
  const float* qb   = (const float*)d_in[3];
  const float* kW   = (const float*)d_in[4];
  const float* kb_  = (const float*)d_in[5];
  const float* vW   = (const float*)d_in[6];
  const float* vb_  = (const float*)d_in[7];
  const float* inqW = (const float*)d_in[8];
  const float* inqb = (const float*)d_in[9];
  const float* inkW = (const float*)d_in[10];
  const float* inkb = (const float*)d_in[11];
  const float* invW = (const float*)d_in[12];
  const float* invb = (const float*)d_in[13];
  const float* outW = (const float*)d_in[14];
  const float* outb = (const float*)d_in[15];
  const float* ln1g = (const float*)d_in[16];
  const float* ln1b = (const float*)d_in[17];
  const float* d1W  = (const float*)d_in[18];
  const float* d1b  = (const float*)d_in[19];
  const float* ln2g = (const float*)d_in[20];
  const float* ln2b = (const float*)d_in[21];
  // d_in[22] = graph_batch (unused by the reference math)
  const unsigned char* mask = (const unsigned char*)d_in[23];  // bool bytes
  float* out = (float*)d_out;

  const size_t MQ = (size_t)CB * CNQ;   // 8192
  const size_t MK = (size_t)CB * CNK;   // 16384

  char* ws = (char*)d_ws;
  size_t off = 0;
  auto alloc = [&](size_t bytes) -> void* {
    void* p = ws + off;
    off += (bytes + 255) & ~(size_t)255;
    return p;
  };

  _Float16* nodes_h = (_Float16*)alloc(MQ * CD * 2);   // later reused as ctx
  _Float16* cond_h  = (_Float16*)alloc(MK * CL * 2);
  _Float16* Wqt  = (_Float16*)alloc((size_t)CD * CD * 2);
  _Float16* Wkt  = (_Float16*)alloc((size_t)CD * CL * 2);
  _Float16* Wvt  = (_Float16*)alloc((size_t)CD * CL * 2);
  _Float16* Wot  = (_Float16*)alloc((size_t)CD * CD * 2);
  _Float16* Wd1t = (_Float16*)alloc((size_t)CD * CD * 2);
  _Float16* qW_h  = (_Float16*)alloc((size_t)CD * CD * 2);
  _Float16* kW_h  = (_Float16*)alloc((size_t)CL * CD * 2);
  _Float16* vW_h  = (_Float16*)alloc((size_t)CL * CD * 2);
  _Float16* inqWt = (_Float16*)alloc((size_t)CD * CD * 2);
  _Float16* inkWt = (_Float16*)alloc((size_t)CD * CD * 2);
  _Float16* invWt = (_Float16*)alloc((size_t)CD * CD * 2);
  float* bq = (float*)alloc(CD * 4);
  float* bk = (float*)alloc(CD * 4);
  float* bv = (float*)alloc(CD * 4);
  _Float16* qh  = (_Float16*)alloc(MQ * CD * 2);   // later reused as x_f16
  _Float16* kh  = (_Float16*)alloc(MK * CD * 2);   // later reused as attn_out f32
  _Float16* vh  = (_Float16*)alloc(MK * CD * 2);   // later reused as x_f32
  _Float16* vhT = (_Float16*)alloc(MK * CD * 2);   // later reused as d1out f32

  _Float16* ctx   = nodes_h;        // nodes_h dead after qh GEMM
  float* attn_out = (float*)kh;     // kh dead after attention (same byte size)
  float* x_f32    = (float*)vh;     // vh dead after transpose
  _Float16* x_f16 = qh;             // qh dead after attention
  float* d1out    = (float*)vhT;    // vhT dead after attention

  // 1) activation converts
  {
    long long n1 = (long long)(MQ * CD);
    cvt_f16_kernel<<<(unsigned)((n1 + 255) / 256), 256, 0, stream>>>(gn, nodes_h, n1);
    long long n2 = (long long)(MK * CL);
    cvt_f16_kernel<<<(unsigned)((n2 + 255) / 256), 256, 0, stream>>>(cond, cond_h, n2);
  }
  // 2) weight converts (+ transposed copies of the right-hand operands)
  cvt_f16_kernel<<<(CD * CD + 255) / 256, 256, 0, stream>>>(qW, qW_h, (long long)CD * CD);
  cvt_f16_kernel<<<(CL * CD + 255) / 256, 256, 0, stream>>>(kW, kW_h, (long long)CL * CD);
  cvt_f16_kernel<<<(CL * CD + 255) / 256, 256, 0, stream>>>(vW, vW_h, (long long)CL * CD);
  cvt_tr_kernel<<<(CD * CD + 255) / 256, 256, 0, stream>>>(inqW, inqWt, CD, CD);
  cvt_tr_kernel<<<(CD * CD + 255) / 256, 256, 0, stream>>>(inkW, inkWt, CD, CD);
  cvt_tr_kernel<<<(CD * CD + 255) / 256, 256, 0, stream>>>(invW, invWt, CD, CD);
  cvt_tr_kernel<<<(CD * CD + 255) / 256, 256, 0, stream>>>(outW, Wot, CD, CD);
  cvt_tr_kernel<<<(CD * CD + 255) / 256, 256, 0, stream>>>(d1W, Wd1t, CD, CD);

  // 3) fuse outer projection with MHA in-proj via WMMA (EP=3: transposed store)
  gemm_kernel<3><<<dim3(CD / 128, CD / 128), 256, 0, stream>>>(
      qW_h, inqWt, nullptr, nullptr, Wqt, nullptr, CD, CD, CD);
  gemm_kernel<3><<<dim3(CD / 128, CL / 128), 256, 0, stream>>>(
      kW_h, inkWt, nullptr, nullptr, Wkt, nullptr, CL, CD, CD);
  gemm_kernel<3><<<dim3(CD / 128, CL / 128), 256, 0, stream>>>(
      vW_h, invWt, nullptr, nullptr, Wvt, nullptr, CL, CD, CD);
  fuse_b_kernel<<<(CD + 255) / 256, 256, 0, stream>>>(qb, inqW, inqb, bq, CD, CD);
  fuse_b_kernel<<<(CD + 255) / 256, 256, 0, stream>>>(kb_, inkW, inkb, bk, CD, CD);
  fuse_b_kernel<<<(CD + 255) / 256, 256, 0, stream>>>(vb_, invW, invb, bv, CD, CD);

  // 4) fused projections (WMMA GEMMs)
  gemm_kernel<0><<<dim3(CD / 128, (unsigned)(MQ / 128)), 256, 0, stream>>>(
      nodes_h, Wqt, bq, nullptr, qh, nullptr, (int)MQ, CD, CD);
  gemm_kernel<0><<<dim3(CD / 128, (unsigned)(MK / 128)), 256, 0, stream>>>(
      cond_h, Wkt, bk, nullptr, kh, nullptr, (int)MK, CD, CL);
  gemm_kernel<0><<<dim3(CD / 128, (unsigned)(MK / 128)), 256, 0, stream>>>(
      cond_h, Wvt, bv, nullptr, vh, nullptr, (int)MK, CD, CL);

  // 5) V transpose -> key-contiguous B-fragment layout
  {
    long long n = (long long)CB * CH * CDH * CNK;
    vt_kernel<<<(unsigned)((n + 255) / 256), 256, 0, stream>>>(vh, vhT);
  }

  // 6) flash attention (6144 waves / 768 blocks)
  attn_kernel<<<(CB * CH * (CNQ / 16)) / 8, 256, 0, stream>>>(qh, kh, vhT, mask, ctx);

  // 7) out-proj -> f32
  gemm_kernel<1><<<dim3(CD / 128, (unsigned)(MQ / 128)), 256, 0, stream>>>(
      ctx, Wot, outb, nullptr, nullptr, attn_out, (int)MQ, CD, CD);

  // 8) LN1(attn_out + graph_nodes) -> x (f32 residual + f16 GEMM input)
  ln_kernel<<<(unsigned)MQ, 256, 0, stream>>>(attn_out, gn, ln1g, ln1b, x_f32, x_f16, CD);

  // 9) d1 GEMM with fused leaky_relu + residual
  gemm_kernel<2><<<dim3(CD / 128, (unsigned)(MQ / 128)), 256, 0, stream>>>(
      x_f16, Wd1t, d1b, x_f32, nullptr, d1out, (int)MQ, CD, CD);

  // 10) LN2 -> final output
  ln_kernel<<<(unsigned)MQ, 256, 0, stream>>>(d1out, nullptr, ln2g, ln2b, out, nullptr, CD);
}
